// GraphEmbedding_25752623907452
// MI455X (gfx1250) — matrix operand order repaired and checked by reading, not compiled, not run on hardware
//
#include <hip/hip_runtime.h>
#include <hip/hip_bf16.h>
#include <math.h>

// ---------------------------------------------------------------------------
// TGN temporal attention layer for MI455X (gfx1250, wave32, WMMA bf16).
//
// Algebraic refactor: instead of materializing k,v = kv_in @ Wk/Wv
// ([163840,768]@[768,512], ~258 GFLOP + 672MB of k/v traffic), fold the
// projections into the query / context sides (K=20 neighbors << dh=256):
//   scores[b,h,kn] = qk[b,h,:] . kv_row      with qk = Wk-projected query
//   ctx[b,h,:]     = (sum_kn attn*kv_row) @ Wv[:,h]
// Dense-GEMM work drops to ~27 GFLOP (all v_wmma_f32_16x16x32_bf16), kv_in is
// consumed exactly once inside a fused gather+attention kernel, so runtime is
// set by the gather floor (~0.6GB @ 23.3 TB/s ~= 30-50us).
//
// GEMM: 64x128 workgroup tile, 4 waves (2x2), wave tile 32x64 = 8 WMMA per
// k-step, LDS double-buffered so next tile's global loads overlap compute.
// ---------------------------------------------------------------------------

#define B_DIM   8192
#define K_NBR   20
#define D_FEAT  256
#define DQ_DIM  512
#define DK_DIM  768
#define KVS     769          // padded LDS row stride (bank-conflict free)

typedef __attribute__((ext_vector_type(16))) __bf16 v16bf;
typedef __attribute__((ext_vector_type(8)))  __bf16 v8bf;
typedef __attribute__((ext_vector_type(8)))  float  v8f;

// ---------------------------------------------------------------------------
// Generic transpose + f32->bf16 weight conversion:  dst[n*Kd + k] = src[k*rs + n*cs]
// ---------------------------------------------------------------------------
__global__ __launch_bounds__(256)
void transpose_to_bf16(const float* __restrict__ src, __bf16* __restrict__ dst,
                       int N, int Kd, int rs, int cs)
{
    int idx = blockIdx.x * 256 + threadIdx.x;
    if (idx >= N * Kd) return;
    int n = idx / Kd, k = idx - n * Kd;
    dst[idx] = (__bf16)src[(size_t)k * rs + (size_t)n * cs];
}

// ---------------------------------------------------------------------------
// query[b] = [ memory[s]+node_table[s] , cos(0*w + b) ]  (bf16)
// also drops src_feat (bf16) into the fc1 concat buffer, cols 512..767
// ---------------------------------------------------------------------------
__global__ __launch_bounds__(256)
void build_query(const int* __restrict__ src_nodes,
                 const float* __restrict__ memory,
                 const float* __restrict__ node_table,
                 const float* __restrict__ time_b,
                 __bf16* __restrict__ query_bf,
                 __bf16* __restrict__ fc1A)
{
    int b = blockIdx.x;
    int t = threadIdx.x;                       // 0..255 (== D_FEAT)
    int s = src_nodes[b];
    float sf = memory[(size_t)s * D_FEAT + t] + node_table[(size_t)s * D_FEAT + t];
    query_bf[(size_t)b * DQ_DIM + t]          = (__bf16)sf;
    query_bf[(size_t)b * DQ_DIM + D_FEAT + t] = (__bf16)__cosf(time_b[t]);
    fc1A[(size_t)b * (DQ_DIM + D_FEAT) + DQ_DIM + t] = (__bf16)sf;
}

// ---------------------------------------------------------------------------
// bf16 WMMA GEMM: C[M,N] = A[M,K] @ B[K,N], B supplied pre-transposed BT[N,K].
// Grid: (M/64, N/128). Block: 128 threads = 4 waves (2x2), wave tile 32x64,
// 2x4 v_wmma_f32_16x16x32_bf16 per wave per 32-deep K step, double-buffered.
// M mult of 64, N mult of 128, K mult of 32 (dims here: 256/512/768/8192).
// ---------------------------------------------------------------------------
template<bool OUT_BF16, bool RELU, bool HAS_BIAS>
__global__ __launch_bounds__(128)
void gemm_bf16(const __bf16* __restrict__ A,  int lda,
               const __bf16* __restrict__ BT, int ldb,
               void* __restrict__ Cout, int ldc,
               const float* __restrict__ bias, int K)
{
    constexpr int STR = 40;                    // 32 + 8 pad (80B rows, 16B aligned)
    __shared__ __bf16 As[2][64][STR];
    __shared__ __bf16 Bs[2][128][STR];

    const int t  = threadIdx.x;
    const int m0 = blockIdx.x * 64;
    const int n0 = blockIdx.y * 128;
    const int w  = t >> 5, l = t & 31;
    const int wm = (w >> 1) * 32;              // wave row offset in tile
    const int wn = (w & 1)  * 64;              // wave col offset in tile
    const int lrow   = l & 15;
    const int khalfA = (l < 16) ? 0 : 8;       // ISA 16-bit A layout
    const int khalfB = (l < 16) ? 0 : 8;       // ISA 16-bit B layout

    v8f acc[2][4];
    #pragma unroll
    for (int mi = 0; mi < 2; ++mi)
        #pragma unroll
        for (int ni = 0; ni < 4; ++ni)
            #pragma unroll
            for (int e = 0; e < 8; ++e) acc[mi][ni][e] = 0.0f;

    // cooperative loader: A 64x32 (16 el/thr), B 128x32 (32 el/thr)
    const int lr0 = (t * 8) >> 5;              // 0..31
    const int lc0 = (t * 8) & 31;              // 0,8,16,24

    auto loadTile = [&](int buf, int kk) {
        const __bf16* sa = A + (size_t)(m0 + lr0) * lda + kk + lc0;
        *(v8bf*)&As[buf][lr0]     [lc0] = *(const v8bf*)sa;
        *(v8bf*)&As[buf][lr0 + 32][lc0] = *(const v8bf*)(sa + (size_t)32 * lda);
        const __bf16* sb = BT + (size_t)(n0 + lr0) * ldb + kk + lc0;
        #pragma unroll
        for (int r = 0; r < 4; ++r)
            *(v8bf*)&Bs[buf][lr0 + 32 * r][lc0] =
                *(const v8bf*)(sb + (size_t)(32 * r) * ldb);
    };

    loadTile(0, 0);
    int cur = 0;
    for (int kk = 0; kk < K; kk += 32) {
        __syncthreads();                       // tile `cur` ready; prev compute done
        if (kk + 32 < K) loadTile(cur ^ 1, kk + 32);   // prefetch next tile

        v16bf af[2], bfr[4];
        #pragma unroll
        for (int mi = 0; mi < 2; ++mi) {       // A frag: K{base..+7, base+16..+23}
            int r = wm + mi * 16 + lrow;
            v8bf lo = *(const v8bf*)&As[cur][r][khalfA];
            v8bf hi = *(const v8bf*)&As[cur][r][khalfA + 16];
            #pragma unroll
            for (int e = 0; e < 8; ++e) { af[mi][e] = lo[e]; af[mi][8 + e] = hi[e]; }
        }
        #pragma unroll
        for (int ni = 0; ni < 4; ++ni) {       // B frag: K{base..+7, base+16..+23}
            int n = wn + ni * 16 + lrow;
            v8bf lo = *(const v8bf*)&Bs[cur][n][khalfB];
            v8bf hi = *(const v8bf*)&Bs[cur][n][khalfB + 16];
            #pragma unroll
            for (int e = 0; e < 8; ++e) { bfr[ni][e] = lo[e]; bfr[ni][8 + e] = hi[e]; }
        }
        #pragma unroll
        for (int mi = 0; mi < 2; ++mi)
            #pragma unroll
            for (int ni = 0; ni < 4; ++ni)
                acc[mi][ni] = __builtin_amdgcn_wmma_f32_16x16x32_bf16(
                    false, af[mi], false, bfr[ni], (short)0, acc[mi][ni], false, false);
        cur ^= 1;
    }

    // epilogue: C/D layout -> VGPR i holds M = i + (lane>=16 ? 8 : 0), N = lane%16
    #pragma unroll
    for (int mi = 0; mi < 2; ++mi)
        #pragma unroll
        for (int ni = 0; ni < 4; ++ni) {
            int rbase = m0 + wm + mi * 16 + ((l < 16) ? 0 : 8);
            int c     = n0 + wn + ni * 16 + lrow;
            float bv  = HAS_BIAS ? bias[c] : 0.0f;
            #pragma unroll
            for (int i = 0; i < 8; ++i) {
                float v = acc[mi][ni][i] + bv;
                if (RELU) v = fmaxf(v, 0.0f);
                size_t idx = (size_t)(rbase + i) * ldc + c;
                if (OUT_BF16) ((__bf16*)Cout)[idx] = (__bf16)v;
                else          ((float*)Cout)[idx]  = v;
            }
        }
}

// ---------------------------------------------------------------------------
// Fused gather + temporal attention (per source row b, 256 threads):
//   build kv rows [20,768] in LDS (gathers + cos time-encode),
//   scores = qk . kv (wave-strided dots + shfl reduce), masked softmax,
//   wctx[h,:] = sum_kn attn * kv  -> bf16 workspace.
// ---------------------------------------------------------------------------
__global__ __launch_bounds__(256)
void attention_fused(const float* __restrict__ timestamps,
                     const int*   __restrict__ neighbors,
                     const int*   __restrict__ edge_idxs,
                     const float* __restrict__ edge_times,
                     const float* __restrict__ memory,
                     const float* __restrict__ node_table,
                     const float* __restrict__ edge_features,
                     const float* __restrict__ time_w,
                     const float* __restrict__ time_b,
                     const __bf16* __restrict__ qk,
                     __bf16* __restrict__ wctx)
{
    __shared__ float kv[K_NBR * KVS];          // 20 x 769 f32, conflict-free stride
    __shared__ float qs[2 * DK_DIM];
    __shared__ float sc[2][K_NBR];
    __shared__ float at[2][K_NBR];
    __shared__ int   nmask[K_NBR];

    const int b = blockIdx.x;
    const int t = threadIdx.x;

    for (int j = t; j < 2 * DK_DIM; j += 256)
        qs[j] = (float)qk[(size_t)b * 2 * DK_DIM + j];
    const float ts = timestamps[b];

    for (int kn = 0; kn < K_NBR; ++kn) {
        int   nbr = neighbors[b * K_NBR + kn];
        int   eix = edge_idxs[b * K_NBR + kn];
        float dt  = ts - edge_times[b * K_NBR + kn];
        if (t == 0) nmask[kn] = (nbr == 0);
        for (int j = t; j < DK_DIM; j += 256) {
            float v;
            if (j < 256)      v = memory[(size_t)nbr * 256 + j] + node_table[(size_t)nbr * 256 + j];
            else if (j < 512) { int j2 = j - 256; v = __cosf(dt * time_w[j2] + time_b[j2]); }
            else              v = edge_features[(size_t)eix * 256 + (j - 512)];
            kv[kn * KVS + j] = v;
        }
    }
    __syncthreads();

    // 40 (kn,h) dot products over 8 waves (5 each), shfl_xor reduce per wave
    const int w = t >> 5, l = t & 31;
    for (int p = w; p < 2 * K_NBR; p += 8) {
        int kn = p >> 1, h = p & 1;
        float s = 0.0f;
        for (int j = l; j < DK_DIM; j += 32) s += kv[kn * KVS + j] * qs[h * DK_DIM + j];
        #pragma unroll
        for (int off = 16; off > 0; off >>= 1) s += __shfl_xor(s, off, 32);
        if (l == 0) sc[h][kn] = s * 0.0625f;   // 1/sqrt(dh=256)
    }
    __syncthreads();

    if (t < 2) {                               // masked softmax per head
        int h = t;
        float v[K_NBR], mx = -1e30f;
        bool invalid = true;
        #pragma unroll
        for (int kn = 0; kn < K_NBR; ++kn) {
            float s = nmask[kn] ? -1e10f : sc[h][kn];
            invalid = invalid && (nmask[kn] != 0);
            v[kn] = s; mx = fmaxf(mx, s);
        }
        float sum = 0.0f;
        #pragma unroll
        for (int kn = 0; kn < K_NBR; ++kn) { float e = __expf(v[kn] - mx); v[kn] = e; sum += e; }
        float inv = invalid ? 0.0f : 1.0f / sum;   // invalid row -> zero ctx (TGN)
        #pragma unroll
        for (int kn = 0; kn < K_NBR; ++kn) at[h][kn] = v[kn] * inv;
    }
    __syncthreads();

    for (int idx = t; idx < 2 * DK_DIM; idx += 256) {
        int h = idx / DK_DIM, j = idx - h * DK_DIM;
        float s = 0.0f;
        #pragma unroll
        for (int kn = 0; kn < K_NBR; ++kn) s += at[h][kn] * kv[kn * KVS + j];
        wctx[(size_t)b * 2 * DK_DIM + idx] = (__bf16)s;
    }
}

// ---------------------------------------------------------------------------
extern "C" void kernel_launch(void* const* d_in, const int* in_sizes, int n_in,
                              void* d_out, int out_size, void* d_ws, size_t ws_size,
                              hipStream_t stream)
{
    (void)in_sizes; (void)n_in; (void)out_size; (void)ws_size;

    const int*   source_nodes  = (const int*)  d_in[0];
    const float* timestamps    = (const float*)d_in[1];
    const int*   neighbors     = (const int*)  d_in[2];
    const int*   edge_idxs     = (const int*)  d_in[3];
    const float* edge_times    = (const float*)d_in[4];
    const float* memory        = (const float*)d_in[5];
    const float* node_table    = (const float*)d_in[6];
    const float* edge_features = (const float*)d_in[7];
    const float* time_w        = (const float*)d_in[8];
    const float* time_b        = (const float*)d_in[9];
    const float* Wq            = (const float*)d_in[10];
    const float* Wk            = (const float*)d_in[11];
    const float* Wv            = (const float*)d_in[12];
    const float* Wo            = (const float*)d_in[13];
    const float* W1            = (const float*)d_in[14];
    const float* b1            = (const float*)d_in[15];
    const float* W2            = (const float*)d_in[16];
    const float* b2            = (const float*)d_in[17];

    char* ws = (char*)d_ws;
    size_t off = 0;
    auto carve = [&](size_t bytes) -> void* {
        void* p = ws + off;
        off = (off + bytes + 255) & ~(size_t)255;
        return p;
    };

    const size_t B = B_DIM;
    __bf16* query_bf = (__bf16*)carve(B * DQ_DIM * 2);           // [B,512]
    __bf16* q_bf     = (__bf16*)carve(B * DQ_DIM * 2);           // [B,512]
    __bf16* qk_bf    = (__bf16*)carve(B * 2 * DK_DIM * 2);       // [B,2,768]
    __bf16* wctx_bf  = (__bf16*)carve(B * 2 * DK_DIM * 2);       // [B,2,768]
    __bf16* ctx_bf   = (__bf16*)carve(B * DQ_DIM * 2);           // [B,512]
    __bf16* fc1A     = (__bf16*)carve(B * (DQ_DIM + D_FEAT) * 2);// [B,768]
    __bf16* h_bf     = (__bf16*)carve(B * D_FEAT * 2);           // [B,256]
    __bf16* WqT  = (__bf16*)carve(512 * 512 * 2);
    __bf16* WkT0 = (__bf16*)carve(768 * 256 * 2);
    __bf16* WkT1 = (__bf16*)carve(768 * 256 * 2);
    __bf16* WvT0 = (__bf16*)carve(256 * 768 * 2);
    __bf16* WvT1 = (__bf16*)carve(256 * 768 * 2);
    __bf16* WoT  = (__bf16*)carve(512 * 512 * 2);
    __bf16* W1T  = (__bf16*)carve(256 * 768 * 2);
    __bf16* W2T  = (__bf16*)carve(256 * 256 * 2);
    // total ws use ~96 MB

    auto tgrid = [](int n) { return dim3((n + 255) / 256); };

    // weight prep: dst[n*Kd+k] = src[k*rs + n*cs]  (BT layout for the GEMM)
    transpose_to_bf16<<<tgrid(512*512), 256, 0, stream>>>(Wq,       WqT,  512, 512, 512,   1);
    transpose_to_bf16<<<tgrid(768*256), 256, 0, stream>>>(Wk,       WkT0, 768, 256,   1, 512);
    transpose_to_bf16<<<tgrid(768*256), 256, 0, stream>>>(Wk + 256, WkT1, 768, 256,   1, 512);
    transpose_to_bf16<<<tgrid(256*768), 256, 0, stream>>>(Wv,       WvT0, 256, 768, 512,   1);
    transpose_to_bf16<<<tgrid(256*768), 256, 0, stream>>>(Wv + 256, WvT1, 256, 768, 512,   1);
    transpose_to_bf16<<<tgrid(512*512), 256, 0, stream>>>(Wo,       WoT,  512, 512, 512,   1);
    transpose_to_bf16<<<tgrid(256*768), 256, 0, stream>>>(W1,       W1T,  256, 768, 256,   1);
    transpose_to_bf16<<<tgrid(256*256), 256, 0, stream>>>(W2,       W2T,  256, 256, 256,   1);

    // query build (also writes src_feat into fc1A cols 512..767)
    build_query<<<dim3(B_DIM), 256, 0, stream>>>(source_nodes, memory, node_table,
                                                 time_b, query_bf, fc1A);

    dim3 blk(128);
    // q = query @ Wq                         [8192,512] = [8192,512]@[512,512]
    gemm_bf16<true,  false, false><<<dim3(B_DIM/64, 512/128), blk, 0, stream>>>(
        query_bf, 512, WqT, 512, q_bf, 512, nullptr, 512);
    // qk_h = q_h @ Wk_h^T                    [8192,768] = [8192,256]@[256,768]
    gemm_bf16<true,  false, false><<<dim3(B_DIM/64, 768/128), blk, 0, stream>>>(
        q_bf,       512, WkT0, 256, qk_bf,          2*DK_DIM, nullptr, 256);
    gemm_bf16<true,  false, false><<<dim3(B_DIM/64, 768/128), blk, 0, stream>>>(
        q_bf + 256, 512, WkT1, 256, qk_bf + DK_DIM, 2*DK_DIM, nullptr, 256);

    // fused gather + attention -> wctx (bandwidth-bound, the true floor)
    attention_fused<<<dim3(B_DIM), 256, 0, stream>>>(
        timestamps, neighbors, edge_idxs, edge_times, memory, node_table,
        edge_features, time_w, time_b, qk_bf, wctx_bf);

    // ctx_h = wctx_h @ Wv[:,h]               [8192,256] = [8192,768]@[768,256]
    gemm_bf16<true,  false, false><<<dim3(B_DIM/64, 256/128), blk, 0, stream>>>(
        wctx_bf,          2*DK_DIM, WvT0, 768, ctx_bf,       512, nullptr, 768);
    gemm_bf16<true,  false, false><<<dim3(B_DIM/64, 256/128), blk, 0, stream>>>(
        wctx_bf + DK_DIM, 2*DK_DIM, WvT1, 768, ctx_bf + 256, 512, nullptr, 768);

    // fc1A[:,0:512] = ctx @ Wo (invalid rows already zero via wctx)
    gemm_bf16<true,  false, false><<<dim3(B_DIM/64, 512/128), blk, 0, stream>>>(
        ctx_bf, 512, WoT, 512, fc1A, 768, nullptr, 512);

    // h = relu(fc1A @ W1 + b1)               [8192,256] = [8192,768]@[768,256]
    gemm_bf16<true,  true,  true ><<<dim3(B_DIM/64, 256/128), blk, 0, stream>>>(
        fc1A, 768, W1T, 768, h_bf, 256, b1, 768);

    // out = h @ W2 + b2 (f32)                [8192,256] = [8192,256]@[256,256]
    gemm_bf16<false, false, true ><<<dim3(B_DIM/64, 256/128), blk, 0, stream>>>(
        h_bf, 256, W2T, 256, d_out, 256, b2, 256);
}
